// Affine_83811991814656
// MI455X (gfx1250) — matrix operand order repaired and checked by reading, not compiled, not run on hardware
//
#include <hip/hip_runtime.h>

// ---------------------------------------------------------------------------
// MoE dispatch/combine on CDNA5 (gfx1250, wave32) via bf16x3 split-precision
// WMMA (V_WMMA_F32_16X16X32_BF16):
//   out[t,o] = sum_e g[t,e] * (x[t,:] . W[e,o,:]) + sum_e g[t,e]*b[e,o]
// f32 operands are split v = v_hi + v_lo (bf16 truncation split);
// x.w ~= xhi*whi + xhi*wlo + xlo*whi  (~2^-16 rel err, f32 accumulate).
// Gates applied in f32 at the per-expert K-block epilogue (exact).
//
// Fast path (needs 80MB workspace, L2-resident): one memory-bound pass
// pre-splits x and W into bf16 hi/lo planes, so the GEMM stages pure bf16
// tiles (no per-tile conversion VALU, half the staging bytes).
// Fallback path: fused on-the-fly split (no workspace requirement).
// ---------------------------------------------------------------------------

typedef __attribute__((ext_vector_type(8)))  float  v8f;
typedef __attribute__((ext_vector_type(8)))  __bf16 v8bf;
typedef __attribute__((ext_vector_type(16))) __bf16 v16bf;

#define DIN   1024
#define DOUT  1024
#define NEXP  4
#define BM    64      // tokens per workgroup tile
#define BN    64      // outputs per workgroup tile
#define BK    64      // K-block staged in LDS
#define THREADS 128   // 4 waves (wave32)
#define LDSW  72      // row stride in bf16: 144B rows -> 16B-aligned b128 frags

// Truncation split of two f32 into packed bf16 hi / lo pairs.
__device__ __forceinline__ void split2(float a, float b,
                                       unsigned& hi, unsigned& lo) {
    const unsigned ua = __float_as_uint(a);
    const unsigned ub = __float_as_uint(b);
    const float ra = a - __uint_as_float(ua & 0xFFFF0000u);
    const float rb = b - __uint_as_float(ub & 0xFFFF0000u);
    hi = (ua >> 16) | (ub & 0xFFFF0000u);
    lo = (__float_as_uint(ra) >> 16) | (__float_as_uint(rb) & 0xFFFF0000u);
}

// Build a v16bf A/B fragment from two 8-element (16B) LDS runs.
__device__ __forceinline__ v16bf frag16(const unsigned short* p0,
                                        const unsigned short* p1) {
    const v8bf a = *reinterpret_cast<const v8bf*>(p0);
    const v8bf b = *reinterpret_cast<const v8bf*>(p1);
    return __builtin_shufflevector(a, b, 0, 1, 2, 3, 4, 5, 6, 7,
                                         8, 9, 10, 11, 12, 13, 14, 15);
}

// ---------------------------------------------------------------------------
// Pass 1 (fast path): split f32 array into bf16 hi/lo planes. Memory-bound.
// ---------------------------------------------------------------------------
__global__ __launch_bounds__(256)
void split_planes_kernel(const float4* __restrict__ src,
                         uint2* __restrict__ dhi,   // 4 bf16 per uint2
                         uint2* __restrict__ dlo,
                         int n4) {
    const int i = blockIdx.x * 256 + threadIdx.x;
    if (i < n4) {
        const float4 v = src[i];
        uint2 hi, lo;
        split2(v.x, v.y, hi.x, lo.x);
        split2(v.z, v.w, hi.y, lo.y);
        dhi[i] = hi;
        dlo[i] = lo;
    }
}

// ---------------------------------------------------------------------------
// Pass 2 (fast path): GEMM over pre-split bf16 planes.
// ---------------------------------------------------------------------------
__global__ __launch_bounds__(THREADS)
void moe_wmma_bf16x3_pre(const unsigned short* __restrict__ Xhi, // [T, DIN] bf16
                         const unsigned short* __restrict__ Xlo,
                         const unsigned short* __restrict__ Whi, // [NEXP,DOUT,DIN]
                         const unsigned short* __restrict__ Wlo,
                         const float* __restrict__ gates,        // [T, NEXP]
                         const float* __restrict__ bias,         // [NEXP, DOUT]
                         float* __restrict__ out,                // [T, DOUT]
                         int T)
{
    __shared__ unsigned short sXhi[BM][LDSW];
    __shared__ unsigned short sXlo[BM][LDSW];
    __shared__ unsigned short sWhi[BN][LDSW];
    __shared__ unsigned short sWlo[BN][LDSW];

    const int tid  = threadIdx.x;
    const int wave = tid >> 5;
    const int lane = tid & 31;
    const int h    = lane >> 4;
    const int lrow = lane & 15;

    const int oBase = blockIdx.x * BN;
    const int tBase = blockIdx.y * BM;

    v8f tot[4] = {{}, {}, {}, {}};

    const int aRow = wave * 16 + lrow;
    const int t0   = tBase + wave * 16 + 8 * h;

    for (int kb0 = 0; kb0 < DIN; kb0 += BK) {
        // ---- stage bf16 x tile (pure copy, 16B granules) ----
        #pragma unroll
        for (int i = 0; i < 4; ++i) {
            const int idx = tid + i * THREADS;     // 0..511
            const int r   = idx >> 3;
            const int c8  = idx & 7;
            const size_t off = (size_t)(tBase + r) * DIN + kb0 + c8 * 8;
            *reinterpret_cast<uint4*>(&sXhi[r][c8 * 8]) =
                *reinterpret_cast<const uint4*>(Xhi + off);
            *reinterpret_cast<uint4*>(&sXlo[r][c8 * 8]) =
                *reinterpret_cast<const uint4*>(Xlo + off);
        }
        // hint the next K-block of x toward the caches (global_prefetch_b8)
        if (kb0 + BK < DIN) {
            __builtin_prefetch(Xhi + (size_t)(tBase + (tid >> 1)) * DIN +
                                   kb0 + BK + (tid & 1) * 32, 0, 1);
        }

        for (int e = 0; e < NEXP; ++e) {
            // ---- stage bf16 W_e tile ----
            #pragma unroll
            for (int i = 0; i < 4; ++i) {
                const int idx = tid + i * THREADS;
                const int r   = idx >> 3;
                const int c8  = idx & 7;
                const size_t off =
                    ((size_t)e * DOUT + (oBase + r)) * DIN + kb0 + c8 * 8;
                *reinterpret_cast<uint4*>(&sWhi[r][c8 * 8]) =
                    *reinterpret_cast<const uint4*>(Whi + off);
                *reinterpret_cast<uint4*>(&sWlo[r][c8 * 8]) =
                    *reinterpret_cast<const uint4*>(Wlo + off);
            }
            __syncthreads();

            // ---- bf16x3 WMMA: 2 K-steps of 32, 4 column tiles, 3 wmma each ----
            v8f acc[4] = {{}, {}, {}, {}};
            #pragma unroll
            for (int ks = 0; ks < BK; ks += 32) {
                const v16bf ahi = frag16(&sXhi[aRow][ks + 8 * h],
                                         &sXhi[aRow][ks + 16 + 8 * h]);
                const v16bf alo = frag16(&sXlo[aRow][ks + 8 * h],
                                         &sXlo[aRow][ks + 16 + 8 * h]);
                #pragma unroll
                for (int n = 0; n < 4; ++n) {
                    const int bn = n * 16 + lrow;
                    const v16bf bhi = frag16(&sWhi[bn][ks + 8 * h],
                                             &sWhi[bn][ks + 16 + 8 * h]);
                    const v16bf blo = frag16(&sWlo[bn][ks + 8 * h],
                                             &sWlo[bn][ks + 16 + 8 * h]);
                    acc[n] = __builtin_amdgcn_wmma_f32_16x16x32_bf16(
                                 false, ahi, false, bhi, (short)0, acc[n],
                                 false, false);
                    acc[n] = __builtin_amdgcn_wmma_f32_16x16x32_bf16(
                                 false, ahi, false, blo, (short)0, acc[n],
                                 false, false);
                    acc[n] = __builtin_amdgcn_wmma_f32_16x16x32_bf16(
                                 false, alo, false, bhi, (short)0, acc[n],
                                 false, false);
                }
            }

            // ---- gate (f32, exact) folded into totals ----
            #pragma unroll
            for (int r = 0; r < 8; ++r) {
                const float g = gates[(size_t)(t0 + r) * NEXP + e];
                #pragma unroll
                for (int n = 0; n < 4; ++n) tot[n][r] += g * acc[n][r];
            }
            __syncthreads();
        }
    }

    // ---- epilogue: gate-weighted bias, store (C: VGPR r -> M=r+8h, N=lrow) ----
    #pragma unroll
    for (int r = 0; r < 8; ++r) {
        const int t = t0 + r;
        const float4 g = reinterpret_cast<const float4*>(gates)[t];
        #pragma unroll
        for (int n = 0; n < 4; ++n) {
            const int o = oBase + n * 16 + lrow;
            const float bt = g.x * bias[o]
                           + g.y * bias[DOUT + o]
                           + g.z * bias[2 * DOUT + o]
                           + g.w * bias[3 * DOUT + o];
            out[(size_t)t * DOUT + o] = tot[n][r] + bt;
        }
    }
}

// ---------------------------------------------------------------------------
// Fallback: fused on-the-fly split (no workspace requirement).
// ---------------------------------------------------------------------------
__global__ __launch_bounds__(THREADS)
void moe_wmma_bf16x3_fused(const float* __restrict__ x,
                           const float* __restrict__ gates,
                           const float* __restrict__ W,
                           const float* __restrict__ bias,
                           float* __restrict__ out,
                           int T)
{
    __shared__ unsigned short sXhi[BM][LDSW];
    __shared__ unsigned short sXlo[BM][LDSW];
    __shared__ unsigned short sWhi[BN][LDSW];
    __shared__ unsigned short sWlo[BN][LDSW];

    const int tid  = threadIdx.x;
    const int wave = tid >> 5;
    const int lane = tid & 31;
    const int h    = lane >> 4;
    const int lrow = lane & 15;

    const int oBase = blockIdx.x * BN;
    const int tBase = blockIdx.y * BM;

    v8f tot[4] = {{}, {}, {}, {}};

    const float4* x4 = reinterpret_cast<const float4*>(x);
    const float4* W4 = reinterpret_cast<const float4*>(W);

    const int aRow = wave * 16 + lrow;
    const int t0   = tBase + wave * 16 + 8 * h;

    for (int kb0 = 0; kb0 < DIN; kb0 += BK) {
        #pragma unroll
        for (int i = 0; i < 8; ++i) {
            const int idx = tid + i * THREADS;
            const int r   = idx >> 4;
            const int c4  = idx & 15;
            const float4 v =
                x4[(size_t)(tBase + r) * (DIN / 4) + (kb0 >> 2) + c4];
            uint2 hi, lo;
            split2(v.x, v.y, hi.x, lo.x);
            split2(v.z, v.w, hi.y, lo.y);
            *reinterpret_cast<uint2*>(&sXhi[r][c4 * 4]) = hi;
            *reinterpret_cast<uint2*>(&sXlo[r][c4 * 4]) = lo;
        }

        for (int e = 0; e < NEXP; ++e) {
            #pragma unroll
            for (int i = 0; i < 8; ++i) {
                const int idx = tid + i * THREADS;
                const int r   = idx >> 4;
                const int c4  = idx & 15;
                const float4 v = W4[((size_t)e * DOUT * DIN +
                                    (size_t)(oBase + r) * DIN + kb0) / 4 + c4];
                uint2 hi, lo;
                split2(v.x, v.y, hi.x, lo.x);
                split2(v.z, v.w, hi.y, lo.y);
                *reinterpret_cast<uint2*>(&sWhi[r][c4 * 4]) = hi;
                *reinterpret_cast<uint2*>(&sWlo[r][c4 * 4]) = lo;
            }
            __syncthreads();

            v8f acc[4] = {{}, {}, {}, {}};
            #pragma unroll
            for (int ks = 0; ks < BK; ks += 32) {
                const v16bf ahi = frag16(&sXhi[aRow][ks + 8 * h],
                                         &sXhi[aRow][ks + 16 + 8 * h]);
                const v16bf alo = frag16(&sXlo[aRow][ks + 8 * h],
                                         &sXlo[aRow][ks + 16 + 8 * h]);
                #pragma unroll
                for (int n = 0; n < 4; ++n) {
                    const int bn = n * 16 + lrow;
                    const v16bf bhi = frag16(&sWhi[bn][ks + 8 * h],
                                             &sWhi[bn][ks + 16 + 8 * h]);
                    const v16bf blo = frag16(&sWlo[bn][ks + 8 * h],
                                             &sWlo[bn][ks + 16 + 8 * h]);
                    acc[n] = __builtin_amdgcn_wmma_f32_16x16x32_bf16(
                                 false, ahi, false, bhi, (short)0, acc[n],
                                 false, false);
                    acc[n] = __builtin_amdgcn_wmma_f32_16x16x32_bf16(
                                 false, ahi, false, blo, (short)0, acc[n],
                                 false, false);
                    acc[n] = __builtin_amdgcn_wmma_f32_16x16x32_bf16(
                                 false, alo, false, bhi, (short)0, acc[n],
                                 false, false);
                }
            }

            #pragma unroll
            for (int r = 0; r < 8; ++r) {
                const float g = gates[(size_t)(t0 + r) * NEXP + e];
                #pragma unroll
                for (int n = 0; n < 4; ++n) tot[n][r] += g * acc[n][r];
            }
            __syncthreads();
        }
    }

    #pragma unroll
    for (int r = 0; r < 8; ++r) {
        const int t = t0 + r;
        const float4 g = reinterpret_cast<const float4*>(gates)[t];
        #pragma unroll
        for (int n = 0; n < 4; ++n) {
            const int o = oBase + n * 16 + lrow;
            const float bt = g.x * bias[o]
                           + g.y * bias[DOUT + o]
                           + g.z * bias[2 * DOUT + o]
                           + g.w * bias[3 * DOUT + o];
            out[(size_t)t * DOUT + o] = tot[n][r] + bt;
        }
    }
}

extern "C" void kernel_launch(void* const* d_in, const int* in_sizes, int n_in,
                              void* d_out, int out_size, void* d_ws, size_t ws_size,
                              hipStream_t stream) {
    const float* x     = (const float*)d_in[0];   // [B,S,DIN] f32
    const float* gates = (const float*)d_in[1];   // [B,S,NEXP] f32
    const float* W     = (const float*)d_in[2];   // [NEXP,DOUT,DIN] f32
    const float* bias  = (const float*)d_in[3];   // [NEXP,DOUT] f32
    float* out = (float*)d_out;                   // [B,S,DOUT] f32

    const int T = in_sizes[1] / NEXP;             // 16384 tokens
    const size_t nX = (size_t)T * DIN;            // x elements
    const size_t nW = (size_t)NEXP * DOUT * DIN;  // W elements
    const size_t need = (nX + nW) * 2 * sizeof(unsigned short); // hi+lo planes

    dim3 grid(DOUT / BN, T / BM);                 // (16, 256)

    if (ws_size >= need) {
        unsigned short* Xhi = (unsigned short*)d_ws;
        unsigned short* Xlo = Xhi + nX;
        unsigned short* Whi = Xlo + nX;
        unsigned short* Wlo = Whi + nW;

        const int nX4 = (int)(nX / 4), nW4 = (int)(nW / 4);
        split_planes_kernel<<<(nX4 + 255) / 256, 256, 0, stream>>>(
            (const float4*)x, (uint2*)Xhi, (uint2*)Xlo, nX4);
        split_planes_kernel<<<(nW4 + 255) / 256, 256, 0, stream>>>(
            (const float4*)W, (uint2*)Whi, (uint2*)Wlo, nW4);
        moe_wmma_bf16x3_pre<<<grid, THREADS, 0, stream>>>(
            Xhi, Xlo, Whi, Wlo, gates, bias, out, T);
    } else {
        moe_wmma_bf16x3_fused<<<grid, THREADS, 0, stream>>>(
            x, gates, W, bias, out, T);
    }
}
